// ClusteringLayer_63900523430220
// MI455X (gfx1250) — matrix operand order repaired and checked by reading, not compiled, not run on hardware
//
#include <hip/hip_runtime.h>

typedef __attribute__((ext_vector_type(16))) __bf16        v16bf;
typedef __attribute__((ext_vector_type(2)))  __bf16        v2bf;
typedef __attribute__((ext_vector_type(8)))  float         v8f;
typedef __attribute__((ext_vector_type(8)))  unsigned int  v8ui;

#define FEATS          256
#define NCLUST         128
#define CSTRIDE        272   // bf16 elements per cluster row in LDS (256 + 16 pad, keeps 32B alignment)
#define ROWS_PER_BLOCK 128   // 8 waves * 16 rows

__device__ __forceinline__ float hi_to_f32(unsigned int bits16) {
    return __builtin_bit_cast(float, bits16 << 16);    // exact bf16 -> f32 widen
}
// Two f32 -> packed bf16 pair; lowers to a single v_cvt_pk_bf16_f32
__device__ __forceinline__ unsigned int pack_bf16_pair(float x0, float x1) {
    const v2bf h = {(__bf16)x0, (__bf16)x1};
    return __builtin_bit_cast(unsigned int, h);
}

__global__ __launch_bounds__(256)
void ClusteringLayer_kernel(const float* __restrict__ X,
                            const float* __restrict__ C,
                            float* __restrict__ out) {
    __shared__ unsigned short s_ch[NCLUST * CSTRIDE];  // cluster bf16 hi
    __shared__ unsigned short s_cl[NCLUST * CSTRIDE];  // cluster bf16 lo (residual)
    __shared__ float          s_cn[NCLUST];            // ||c_k||^2

    const int tid = threadIdx.x;

    // ---- Stage clusters (128x256 f32) into LDS as bf16 hi/lo, coalesced float4 loads ----
    #pragma unroll 4
    for (int it = 0; it < 32; ++it) {
        const int v4 = tid + it * 256;          // float4 index into C
        const int n  = v4 >> 6;                 // 64 float4 per cluster row
        const int f  = (v4 & 63) << 2;          // feature offset
        const float4 c4 = ((const float4*)C)[v4];

        const unsigned int hp0 = pack_bf16_pair(c4.x, c4.y);
        const unsigned int hp1 = pack_bf16_pair(c4.z, c4.w);
        const unsigned int lp0 = pack_bf16_pair(
            c4.x - hi_to_f32(hp0 << 16 >> 16),   // note: hp0<<16>>16 == low 16 bits
            c4.y - __builtin_bit_cast(float, hp0 & 0xffff0000u));
        const unsigned int lp1 = pack_bf16_pair(
            c4.z - hi_to_f32(hp1 << 16 >> 16),
            c4.w - __builtin_bit_cast(float, hp1 & 0xffff0000u));

        const unsigned int base = n * CSTRIDE + f;          // ushort units, 8B aligned
        *(uint2*)&s_ch[base] = make_uint2(hp0, hp1);
        *(uint2*)&s_cl[base] = make_uint2(lp0, lp1);
    }
    __syncthreads();

    // ---- ||c_k||^2 from reconstructed hi+lo (consistent with split dot product) ----
    if (tid < NCLUST) {
        float s = 0.f;
        const unsigned int base = tid * CSTRIDE;
        #pragma unroll 8
        for (int f = 0; f < FEATS; ++f) {
            const float v = hi_to_f32(s_ch[base + f]) + hi_to_f32(s_cl[base + f]);
            s = fmaf(v, v, s);
        }
        s_cn[tid] = s;
    }
    __syncthreads();

    // ---- Per-wave 16x128 output tile via bf16 split WMMA ----
    const int lane = tid & 31;
    const int wave = tid >> 5;
    const int half = lane >> 4;       // K-half selector per WMMA layout
    const int l16  = lane & 15;       // A: row within tile; B: col within N-tile
    const int rowBase = blockIdx.x * ROWS_PER_BLOCK + wave * 16;
    const float* xrow = X + (rowBase + l16) * FEATS;

    v8f acc[8];
    #pragma unroll
    for (int nt = 0; nt < 8; ++nt)
        acc[nt] = (v8f){0.f, 0.f, 0.f, 0.f, 0.f, 0.f, 0.f, 0.f};

    float xsq = 0.f;
    #pragma unroll 2
    for (int kk = 0; kk < FEATS; kk += 32) {
        // A fragment, contiguous k order (same as B):
        // elements 0..15 <- f = kk + 16*half + 0..15
        float xv[16];
        const int f1 = kk + 16 * half;
        const int f2 = kk + 16 * half + 8;
        ((float4*)xv)[0] = *(const float4*)(xrow + f1);
        ((float4*)xv)[1] = *(const float4*)(xrow + f1 + 4);
        ((float4*)xv)[2] = *(const float4*)(xrow + f2);
        ((float4*)xv)[3] = *(const float4*)(xrow + f2 + 4);

        // Split each f32 into bf16 hi + bf16 residual; one packed convert per
        // pair for hi and for lo; exact hi widens come off the packed bits.
        v8ui ahp, alp;
        #pragma unroll
        for (int p = 0; p < 8; ++p) {
            const float x0 = xv[2 * p];
            const float x1 = xv[2 * p + 1];
            const unsigned int pk = pack_bf16_pair(x0, x1);
            ahp[p] = pk;
            const float h0 = __builtin_bit_cast(float, pk << 16);
            const float h1 = __builtin_bit_cast(float, pk & 0xffff0000u);
            alp[p] = pack_bf16_pair(x0 - h0, x1 - h1);
            xsq = fmaf(x0, x0, xsq);
            xsq = fmaf(x1, x1, xsq);
        }
        const v16bf ah = __builtin_bit_cast(v16bf, ahp);
        const v16bf al = __builtin_bit_cast(v16bf, alp);

        // B fragment, contiguous k order (same as A): element i <- K = i + 16*half, col = l16
        const int fb = kk + 16 * half;
        #pragma unroll
        for (int nt = 0; nt < 8; ++nt) {
            const int n = nt * 16 + l16;
            const v16bf bh = *(const v16bf*)&s_ch[n * CSTRIDE + fb];
            const v16bf bl = *(const v16bf*)&s_cl[n * CSTRIDE + fb];
            acc[nt] = __builtin_amdgcn_wmma_f32_16x16x32_bf16(
                false, ah, false, bh, (short)0, acc[nt], false, false);
            acc[nt] = __builtin_amdgcn_wmma_f32_16x16x32_bf16(
                false, ah, false, bl, (short)0, acc[nt], false, false);
            acc[nt] = __builtin_amdgcn_wmma_f32_16x16x32_bf16(
                false, al, false, bh, (short)0, acc[nt], false, false);
        }
    }

    // ---- Epilogue: d2 -> q -> row-normalize -> store ----
    // lane L holds ||x||^2-half for row l16; pair L and L^16 covers all features
    const float xs_full = xsq + __shfl_xor(xsq, 16, 32);

    float rowinv[8];
    #pragma unroll
    for (int r = 0; r < 8; ++r) {
        // accumulator register r holds row M = r + 8*half (col = nt*16 + l16)
        const float xn = __shfl(xs_full, r + 8 * half, 32);
        float rs = 0.f;
        #pragma unroll
        for (int nt = 0; nt < 8; ++nt) {
            const int n = nt * 16 + l16;
            const float d2 = xn + s_cn[n] - 2.0f * acc[nt][r];
            const float q = 1.0f / (1.0f + d2);   // ALPHA=1: (1+d2)^(-(1+1)/2) = 1/(1+d2)
            acc[nt][r] = q;
            rs += q;
        }
        rs += __shfl_xor(rs, 1, 32);
        rs += __shfl_xor(rs, 2, 32);
        rs += __shfl_xor(rs, 4, 32);
        rs += __shfl_xor(rs, 8, 32);   // stays within the 16-lane group = one row
        rowinv[r] = 1.0f / rs;
    }

    #pragma unroll
    for (int r = 0; r < 8; ++r) {
        const int row = rowBase + r + 8 * half;
        float* o = out + row * NCLUST;
        #pragma unroll
        for (int nt = 0; nt < 8; ++nt)
            o[nt * 16 + l16] = acc[nt][r] * rowinv[r];
    }
}

extern "C" void kernel_launch(void* const* d_in, const int* in_sizes, int n_in,
                              void* d_out, int out_size, void* d_ws, size_t ws_size,
                              hipStream_t stream) {
    (void)in_sizes; (void)n_in; (void)out_size; (void)d_ws; (void)ws_size;
    const float* X = (const float*)d_in[0];   // (16384, 256) f32
    const float* C = (const float*)d_in[1];   // (128, 256) f32
    float* out = (float*)d_out;               // (16384, 128) f32
    dim3 grid(16384 / ROWS_PER_BLOCK);        // 128 blocks
    dim3 block(256);                          // 8 waves
    ClusteringLayer_kernel<<<grid, block, 0, stream>>>(X, C, out);
}
